// MultiHeadAttentionLayer_55851754717765
// MI455X (gfx1250) — matrix operand order repaired and checked by reading, not compiled, not run on hardware
//
#include <hip/hip_runtime.h>

// ---------------------------------------------------------------------------
// MultiHeadAttentionLayer for MI455X (gfx1250), f16 WMMA with f32 accumulation
// B=4, S=1024, H=1024, NH=16, HD=64, SCALE=8
// Fused attention: scores -> softmax -> attn*V staged through LDS, so the
// 256 MB attention matrix is written to HBM exactly once (it is an output).
// Mask is pre-packed to bits (wave32 __ballot) to kill scattered b32 loads.
// ---------------------------------------------------------------------------

typedef __attribute__((ext_vector_type(16))) _Float16 v16h;
typedef __attribute__((ext_vector_type(8)))  _Float16 v8h;
typedef __attribute__((ext_vector_type(8)))  float    v8f;

constexpr int Bn  = 4;
constexpr int Sn  = 1024;
constexpr int Hn  = 1024;
constexpr int NHn = 16;
constexpr int HDn = 64;
constexpr int Mn  = Bn * Sn;            // 4096 rows for the projection GEMMs
constexpr float INV_SCALE = 0.125f;     // 1/sqrt(HD)
constexpr int SC_PITCH = 1028;          // 1024 + 4 floats: LDS bank skew
constexpr int MW = Sn / 32;             // mask words per row (32)

__device__ __forceinline__ v8f wmma_f16(v16h a, v16h b, v8f c) {
  // D = A(16x32 f16) * B(32x16 f16) + C(16x16 f32)
  return __builtin_amdgcn_wmma_f32_16x16x32_f16(false, a, false, b, (short)0, c,
                                                false, false);
}

// A-operand (16x32, f16, row-major source: row m contiguous in k).
// Lane L: m = m0 + L%16, g = L/16; halves 0..15 = k0+g*16 .. +15 contiguous
// (same k order as load_b_f16).
__device__ __forceinline__ v16h load_a_f16(const _Float16* __restrict__ A,
                                           int lda, int m0, int k0) {
  const int lane = threadIdx.x & 31;
  const int g = lane >> 4;
  const _Float16* p = A + (size_t)(m0 + (lane & 15)) * lda + k0 + g * 16;
  v8h lo = *(const v8h*)p;         // 16B aligned
  v8h hi = *(const v8h*)(p + 8);   // 16B aligned
  return __builtin_shufflevector(lo, hi, 0, 1, 2, 3, 4, 5, 6, 7,
                                 8, 9, 10, 11, 12, 13, 14, 15);
}

// B-operand (32x16, f16). Element (k, n) must come from Bsrc[(n0+n)*ldb + k],
// i.e. Bsrc rows are indexed by the OUTPUT column n and contiguous in k.
// For y = x @ W.T this is row-major W directly; for scores it is row-major K.
// Lane L: n = n0 + L%16, holds k = k0 + (L/16)*16 .. +15 contiguous (32B load).
__device__ __forceinline__ v16h load_b_f16(const _Float16* __restrict__ Bsrc,
                                           int ldb, int n0, int k0) {
  const int lane = threadIdx.x & 31;
  const int g = lane >> 4;
  const _Float16* p = Bsrc + (size_t)(n0 + (lane & 15)) * ldb + k0 + g * 16;
  return *(const v16h*)p;  // 32B aligned for our strides/offsets
}

// ---------------------------------------------------------------------------
__global__ void cvt_f32_to_f16(const float* __restrict__ in,
                               _Float16* __restrict__ out, int n) {
  int i = blockIdx.x * blockDim.x + threadIdx.x;
  if (i < n) out[i] = (_Float16)in[i];
}

// Pack mask (int32, nonzero = true) into a bit matrix: wave32 ballot gives one
// 32-bit word per 32 consecutive elements. Grid covers exactly B*S*S threads.
__global__ void mask_pack_bits(const int* __restrict__ in,
                               unsigned* __restrict__ out) {
  const int i = blockIdx.x * blockDim.x + threadIdx.x;
  const unsigned long long bal = __ballot(in[i] != 0);
  if ((threadIdx.x & 31) == 0) out[i >> 5] = (unsigned)bal;
}

// ---------------------------------------------------------------------------
// Projection GEMM: out[m,n] = sum_k A[m,k] * W[n,k] + bias[n]
// M=4096, N=1024, K=1024. Block = 128 threads (4 waves); wave tile = 16x64.
// MODE 0: f16 out row-major [m*1024+n]      (Q, K, Qs, Ks)
// MODE 1: f32 out row-major                 (final x projection)
// MODE 2: f16 out as Vt[B,NH,HD,S]          (V, pre-transposed for attn*V)
// ---------------------------------------------------------------------------
template <int MODE>
__global__ __launch_bounds__(128)
void gemm_proj(const _Float16* __restrict__ A,
               const _Float16* __restrict__ W,
               const float* __restrict__ bias,
               _Float16* __restrict__ outH,
               float* __restrict__ outF) {
  const int wave = threadIdx.x >> 5;
  const int lane = threadIdx.x & 31;
  const int m0 = blockIdx.y * 16;
  const int n0 = blockIdx.x * 256 + wave * 64;

  v8f acc[4] = {};
  for (int k0 = 0; k0 < Hn; k0 += 32) {
    v16h a = load_a_f16(A, Hn, m0, k0);
#pragma unroll
    for (int j = 0; j < 4; ++j) {
      v16h bb = load_b_f16(W, Hn, n0 + j * 16, k0);
      acc[j] = wmma_f16(a, bb, acc[j]);
    }
  }

  const int nl = lane & 15, g = lane >> 4;
#pragma unroll
  for (int j = 0; j < 4; ++j) {
#pragma unroll
    for (int r = 0; r < 8; ++r) {
      const int m = m0 + r + 8 * g;     // C/D layout: VGPR r -> row r + 8*(L/16)
      const int n = n0 + j * 16 + nl;   // lane%16 -> column
      const float v = acc[j][r] + bias[n];
      if (MODE == 0) {
        outH[(size_t)m * Hn + n] = (_Float16)v;
      } else if (MODE == 1) {
        outF[(size_t)m * Hn + n] = v;
      } else {
        const int b = m >> 10, s = m & 1023;
        const int h = n >> 6,  d = n & 63;
        outH[((((size_t)b * NHn + h) * HDn + d) << 10) + s] = (_Float16)v;
      }
    }
  }
}

// ---------------------------------------------------------------------------
// Fused attention for one 16-query tile of one (b,h):
//   phase 1: scores = where(mask, Qs*Ks^T, Q*K^T)/8 -> LDS (each wave: 256 keys)
//   phase 2: softmax rows in LDS, write normalized probs to global (only pass)
//   phase 3: partial attn*V per wave from LDS probs, reduce via LDS -> ctx f16
// ---------------------------------------------------------------------------
__global__ __launch_bounds__(128, 2)
void attn_fused(const _Float16* __restrict__ Qh,
                const _Float16* __restrict__ Kh,
                const _Float16* __restrict__ Qsh,
                const _Float16* __restrict__ Ksh,
                const _Float16* __restrict__ Vt,
                const unsigned* __restrict__ maskBits,
                float* __restrict__ attnOut,
                _Float16* __restrict__ ctx) {
  __shared__ float sc[16][SC_PITCH];    // ~64 KB scores/probs, skewed pitch
  __shared__ float part[4][16][HDn];    // 16 KB partial attn*V accumulators

  const int wave = threadIdx.x >> 5;
  const int lane = threadIdx.x & 31;
  const int nl = lane & 15, g = lane >> 4;
  const int m0 = blockIdx.x * 16;       // query tile within S
  const int bh = blockIdx.y;
  const int b = bh >> 4, h = bh & 15;

  const size_t base = (size_t)b * Sn * Hn + (size_t)h * HDn;
  const _Float16* Qp  = Qh  + base;
  const _Float16* Kp  = Kh  + base;
  const _Float16* Qsp = Qsh + base;
  const _Float16* Ksp = Ksh + base;
  const _Float16* Vp  = Vt + (size_t)bh * HDn * Sn;

  // Warm caches for V (needed in phase 3) while we crunch scores:
  // 128 threads x 8 x 128B = 128 KB = the whole per-head V tile.
#pragma unroll
  for (int i = 0; i < 8; ++i)
    __builtin_prefetch((const char*)Vp + ((size_t)threadIdx.x + (size_t)i * 128) * 128, 0, 1);

  // ---- Phase 1: scores for this wave's 256-key range ----------------------
  v16h aq0  = load_a_f16(Qp,  Hn, m0, 0);
  v16h aq1  = load_a_f16(Qp,  Hn, m0, 32);
  v16h aqs0 = load_a_f16(Qsp, Hn, m0, 0);
  v16h aqs1 = load_a_f16(Qsp, Hn, m0, 32);

  const unsigned* mrow = maskBits + (size_t)b * Sn * MW;  // mask bits [S, S/32]
#pragma unroll 1   // keep register pressure < 256 VGPRs (no vgpr-msb traffic)
  for (int t = 0; t < 4; ++t) {
    const int n0 = wave * 256 + t * 64;
    v8f acc[4] = {}, accs[4] = {};
#pragma unroll
    for (int j = 0; j < 4; ++j) {
      acc[j]  = wmma_f16(aq0,  load_b_f16(Kp,  Hn, n0 + j * 16, 0),  acc[j]);
      acc[j]  = wmma_f16(aq1,  load_b_f16(Kp,  Hn, n0 + j * 16, 32), acc[j]);
      accs[j] = wmma_f16(aqs0, load_b_f16(Ksp, Hn, n0 + j * 16, 0),  accs[j]);
      accs[j] = wmma_f16(aqs1, load_b_f16(Ksp, Hn, n0 + j * 16, 32), accs[j]);
    }
#pragma unroll
    for (int j = 0; j < 4; ++j) {
      const int nb = n0 + j * 16;        // 16-aligned => bits live in one word
#pragma unroll
      for (int r = 0; r < 8; ++r) {
        const int m = r + 8 * g;         // local query row 0..15
        const unsigned wbits = mrow[(size_t)(m0 + m) * MW + (nb >> 5)];
        const float e  = acc[j][r]  * INV_SCALE;
        const float es = accs[j][r] * INV_SCALE;
        sc[m][nb + nl] = ((wbits >> ((nb & 31) + nl)) & 1u) ? es : e;
      }
    }
  }
  __syncthreads();

  // ---- Phase 2: softmax on 16 rows (4 per wave), single global write ------
  float* arow_base = attnOut + (size_t)bh * Sn * Sn;
#pragma unroll
  for (int rr = 0; rr < 4; ++rr) {
    const int row = wave * 4 + rr;
    float v[32];
    float mx = -3.4e38f;
#pragma unroll
    for (int i = 0; i < 32; ++i) {
      v[i] = sc[row][i * 32 + lane];
      mx = fmaxf(mx, v[i]);
    }
#pragma unroll
    for (int off = 16; off > 0; off >>= 1) mx = fmaxf(mx, __shfl_xor(mx, off, 32));
    float s = 0.0f;
#pragma unroll
    for (int i = 0; i < 32; ++i) {
      v[i] = __expf(v[i] - mx);
      s += v[i];
    }
#pragma unroll
    for (int off = 16; off > 0; off >>= 1) s += __shfl_xor(s, off, 32);
    const float inv = 1.0f / s;
    float* arow = arow_base + (size_t)(m0 + row) * Sn;
#pragma unroll
    for (int i = 0; i < 32; ++i) {
      const float pv = v[i] * inv;
      sc[row][i * 32 + lane] = pv;
      arow[i * 32 + lane] = pv;       // the one-and-only HBM pass over attn
    }
  }
  __syncthreads();

  // ---- Phase 3: partial attn*V over this wave's 256-key range -------------
  v8f acc[4] = {};
#pragma unroll 2
  for (int k0 = wave * 256; k0 < wave * 256 + 256; k0 += 32) {
    // A-fragment from LDS probs, f32 -> f16 in-register (ds_load + v_cvt)
    v16h a;
    {
      const int ml = lane & 15;
      const float* p0 = &sc[ml][k0 + g * 16];
      const float* p1 = &sc[ml][k0 + g * 16 + 8];
#pragma unroll
      for (int i = 0; i < 8; ++i) a[i] = (_Float16)p0[i];
#pragma unroll
      for (int i = 0; i < 8; ++i) a[8 + i] = (_Float16)p1[i];
    }
#pragma unroll
    for (int j = 0; j < 4; ++j) {
      v16h bv = load_b_f16(Vp, Sn, j * 16, k0);
      acc[j] = wmma_f16(a, bv, acc[j]);
    }
  }
#pragma unroll
  for (int j = 0; j < 4; ++j) {
#pragma unroll
    for (int r = 0; r < 8; ++r)
      part[wave][r + 8 * g][j * 16 + nl] = acc[j][r];
  }
  __syncthreads();

  // ---- Phase 4: reduce the 4 per-wave partials, emit f16 context ----------
  for (int i = threadIdx.x; i < 16 * HDn; i += 128) {
    const int m = i >> 6, d = i & 63;
    const float sum = part[0][m][d] + part[1][m][d] + part[2][m][d] + part[3][m][d];
    ctx[((size_t)b * Sn + m0 + m) * Hn + h * HDn + d] = (_Float16)sum;
  }
}

// ---------------------------------------------------------------------------
extern "C" void kernel_launch(void* const* d_in, const int* in_sizes, int n_in,
                              void* d_out, int out_size, void* d_ws, size_t ws_size,
                              hipStream_t stream) {
  (void)in_sizes; (void)n_in; (void)out_size; (void)ws_size;

  const float* query  = (const float*)d_in[0];
  const float* key    = (const float*)d_in[1];
  const float* qside  = (const float*)d_in[2];
  const float* kside  = (const float*)d_in[3];
  const float* value  = (const float*)d_in[4];
  const int*   mask   = (const int*)d_in[5];
  const float* Wq = (const float*)d_in[6];  const float* bq = (const float*)d_in[7];
  const float* Wk = (const float*)d_in[8];  const float* bk = (const float*)d_in[9];
  const float* Wqs = (const float*)d_in[10]; const float* bqs = (const float*)d_in[11];
  const float* Wks = (const float*)d_in[12]; const float* bks = (const float*)d_in[13];
  const float* Wv = (const float*)d_in[14]; const float* bv = (const float*)d_in[15];
  const float* Wo = (const float*)d_in[16]; const float* bo = (const float*)d_in[17];

  constexpr size_t IN = (size_t)Mn * Hn;  // 4194304 elements per [B,S,H] tensor
  constexpr size_t WE = (size_t)Hn * Hn;  // 1048576 elements per weight matrix

  _Float16* ws = (_Float16*)d_ws;         // ~105 MB of staging
  _Float16* xq  = ws + 0 * IN;
  _Float16* xk  = ws + 1 * IN;
  _Float16* xqs = ws + 2 * IN;
  _Float16* xks = ws + 3 * IN;
  _Float16* xv  = ws + 4 * IN;
  _Float16* Qh  = ws + 5 * IN;
  _Float16* Kh  = ws + 6 * IN;
  _Float16* Qsh = ws + 7 * IN;
  _Float16* Ksh = ws + 8 * IN;
  _Float16* Vt  = ws + 9 * IN;            // [B,NH,HD,S]
  _Float16* ctx = ws + 10 * IN;           // [B,S,H]
  _Float16* wq  = ws + 11 * IN + 0 * WE;
  _Float16* wk  = ws + 11 * IN + 1 * WE;
  _Float16* wqs = ws + 11 * IN + 2 * WE;
  _Float16* wks = ws + 11 * IN + 3 * WE;
  _Float16* wv  = ws + 11 * IN + 4 * WE;
  _Float16* wo  = ws + 11 * IN + 5 * WE;
  unsigned* maskBits = (unsigned*)(ws + 11 * IN + 6 * WE);  // B*S*S/32 words

  float* outX = (float*)d_out;            // [B,S,H] f32
  float* outA = outX + IN;                // [B,NH,S,S] f32 attention

  // 1) f32 -> f16 staging + mask bit-packing
  const dim3 cb(256);
  auto cvt = [&](const float* src, _Float16* dst, size_t n) {
    cvt_f32_to_f16<<<dim3((unsigned)((n + 255) / 256)), cb, 0, stream>>>(src, dst, (int)n);
  };
  cvt(query, xq, IN);  cvt(key, xk, IN);  cvt(qside, xqs, IN);
  cvt(kside, xks, IN); cvt(value, xv, IN);
  cvt(Wq, wq, WE);  cvt(Wk, wk, WE);  cvt(Wqs, wqs, WE);
  cvt(Wks, wks, WE); cvt(Wv, wv, WE); cvt(Wo, wo, WE);
  mask_pack_bits<<<dim3((unsigned)(((size_t)Bn * Sn * Sn) / 256)), cb, 0, stream>>>(
      mask, maskBits);

  // 2) projections (WMMA)
  const dim3 gblk(128);          // 4 waves
  const dim3 ggrid(4, Mn / 16);  // N/256 x M/16
  gemm_proj<0><<<ggrid, gblk, 0, stream>>>(xq,  wq,  bq,  Qh,  nullptr);
  gemm_proj<0><<<ggrid, gblk, 0, stream>>>(xk,  wk,  bk,  Kh,  nullptr);
  gemm_proj<0><<<ggrid, gblk, 0, stream>>>(xqs, wqs, bqs, Qsh, nullptr);
  gemm_proj<0><<<ggrid, gblk, 0, stream>>>(xks, wks, bks, Ksh, nullptr);
  gemm_proj<2><<<ggrid, gblk, 0, stream>>>(xv,  wv,  bv,  Vt,  nullptr);

  // 3) fused scores + softmax + attn*V (one HBM pass over the S x S matrix)
  attn_fused<<<dim3(Sn / 16, Bn * NHn), gblk, 0, stream>>>(Qh, Kh, Qsh, Ksh, Vt,
                                                           maskBits, outA, ctx);

  // 4) x = ctx @ Wo^T + bo -> f32 d_out
  gemm_proj<1><<<ggrid, gblk, 0, stream>>>(ctx, wo, bo, nullptr, outX);
}
